// ContextuallSelfAttention_26207890440753
// MI455X (gfx1250) — matrix operand order repaired and checked by reading, not compiled, not run on hardware
//
#include <hip/hip_runtime.h>
#include <math.h>

typedef __attribute__((ext_vector_type(16))) __bf16 v16bf;
typedef __attribute__((ext_vector_type(8)))  __bf16 v8bf;
typedef __attribute__((ext_vector_type(8)))  float  v8f;

#define GEMM_K 256   // d == 256 in the reference model

static __device__ inline __bf16 f2bf(float f) {
    union { float f; unsigned u; } v; v.f = f;
    unsigned r = v.u + 0x7FFFu + ((v.u >> 16) & 1u);   // round-to-nearest-even
    unsigned short h = (unsigned short)(r >> 16);
    __bf16 o;
    __builtin_memcpy(&o, &h, 2);
    return o;
}

// Streaming f32 -> bf16 convert (n % 4 == 0).
__global__ void cvt_bf16(const float* __restrict__ in, __bf16* __restrict__ out, long n)
{
    long i = ((long)blockIdx.x * blockDim.x + threadIdx.x) * 4;
    if (i >= n) return;
    float4 v = *(const float4*)(in + i);
    out[i]     = f2bf(v.x);
    out[i + 1] = f2bf(v.y);
    out[i + 2] = f2bf(v.z);
    out[i + 3] = f2bf(v.w);
}

// W[K][N] f32  ->  WT[N][K] bf16  (weights are tiny; one-shot)
__global__ void cvt_transpose(const float* __restrict__ Wsrc, __bf16* __restrict__ Wdst, int N)
{
    int idx = blockIdx.x * blockDim.x + threadIdx.x;    // over N*K
    if (idx >= N * GEMM_K) return;
    int n = idx / GEMM_K, k = idx % GEMM_K;
    Wdst[idx] = f2bf(Wsrc[(long)k * N + n]);
}

// C[M x N] = A[M x 256] @ B[256 x N] + bias, A bf16 row-major, B given transposed bf16 [N][256].
// Block = 256 threads = 8 waves. Wave computes 64 rows x 16 cols (4 WMMA tiles, shared B frag).
// Grid: (Mtot/512, N/16).
__global__ __launch_bounds__(256) void gemm_bf16_wmma(
    const __bf16* __restrict__ A, const __bf16* __restrict__ BT,
    const float* __restrict__ bias, float* __restrict__ C, int N)
{
    const int lane = threadIdx.x & 31;
    const int wave = threadIdx.x >> 5;
    const int half = lane >> 4;      // 0: lanes 0-15, 1: lanes 16-31
    const int l15  = lane & 15;
    const long rowBase = (long)blockIdx.x * 512 + wave * 64;
    const int  col = blockIdx.y * 16 + l15;

    v8f acc0 = {}, acc1 = {}, acc2 = {}, acc3 = {};
    const __bf16* bcol  = BT + (long)col * GEMM_K;
    const __bf16* arow0 = A + (rowBase + l15) * (long)GEMM_K;

#pragma unroll
    for (int kk = 0; kk < GEMM_K; kk += 32) {
        // B fragment: col = l15, K chunks [half*8 .. +7] and [16+half*8 .. +7] in transposed layout.
        const __bf16* bp = bcol + kk + half * 8;
        v8bf b0 = *(const v8bf*)(bp);
        v8bf b1 = *(const v8bf*)(bp + 16);
        v16bf bfrag = __builtin_shufflevector(b0, b1,
            0, 1, 2, 3, 4, 5, 6, 7, 8, 9, 10, 11, 12, 13, 14, 15);

        // A fragment per tile: K chunks [half*8 .. +7] and [16+half*8 .. +7], contiguous b128 loads.
        const __bf16* ap = arow0 + kk + half * 8;
#define A_FRAG(dst, tile)                                                     \
        {                                                                     \
            const __bf16* q = ap + (long)(tile) * 16 * GEMM_K;                \
            v8bf lo = *(const v8bf*)(q);                                      \
            v8bf hi = *(const v8bf*)(q + 16);                                 \
            dst = __builtin_shufflevector(lo, hi,                             \
                0, 1, 2, 3, 4, 5, 6, 7, 8, 9, 10, 11, 12, 13, 14, 15);        \
        }
        v16bf a0, a1, a2, a3;
        A_FRAG(a0, 0) A_FRAG(a1, 1) A_FRAG(a2, 2) A_FRAG(a3, 3)
#undef A_FRAG
        acc0 = __builtin_amdgcn_wmma_f32_16x16x32_bf16(false, a0, false, bfrag, (short)0, acc0, false, false);
        acc1 = __builtin_amdgcn_wmma_f32_16x16x32_bf16(false, a1, false, bfrag, (short)0, acc1, false, false);
        acc2 = __builtin_amdgcn_wmma_f32_16x16x32_bf16(false, a2, false, bfrag, (short)0, acc2, false, false);
        acc3 = __builtin_amdgcn_wmma_f32_16x16x32_bf16(false, a3, false, bfrag, (short)0, acc3, false, false);
    }

    const float bb = bias[col];
#pragma unroll
    for (int t = 0; t < 4; ++t) {
        v8f acc = (t == 0) ? acc0 : (t == 1) ? acc1 : (t == 2) ? acc2 : acc3;
#pragma unroll
        for (int i = 0; i < 8; ++i) {
            // C fragment: VGPR i -> row = half*8 + i, col = l15
            long row = rowBase + t * 16 + (half << 3) + i;
            C[row * (long)N + col] = acc[i] + bb;
        }
    }
}

// In-place: sl[b,q,m,1,p,c] = ref[b,q,0,c] + sl/(c==0?W:H)
__global__ void loc_fix(float* __restrict__ sl, const float* __restrict__ ref,
                        const int* __restrict__ Hp, const int* __restrict__ Wp, long total)
{
    long i = (long)blockIdx.x * blockDim.x + threadIdx.x;
    if (i >= total) return;
    long bq = i >> 6;           // /64  (M*P*2)
    int  c  = (int)(i & 1);
    float norm = (float)(c == 0 ? *Wp : *Hp);
    sl[i] = ref[bq * 2 + c] + sl[i] / norm;
}

// In-place softmax over groups of 4 (P axis).
__global__ void softmax4(float* __restrict__ a, long groups)
{
    long g = (long)blockIdx.x * blockDim.x + threadIdx.x;
    if (g >= groups) return;
    float* p = a + g * 4;
    float x0 = p[0], x1 = p[1], x2 = p[2], x3 = p[3];
    float m = fmaxf(fmaxf(x0, x1), fmaxf(x2, x3));
    float e0 = expf(x0 - m), e1 = expf(x1 - m), e2 = expf(x2 - m), e3 = expf(x3 - m);
    float inv = 1.0f / (e0 + e1 + e2 + e3);
    p[0] = e0 * inv; p[1] = e1 * inv; p[2] = e2 * inv; p[3] = e3 * inv;
}

static __device__ inline float corner_val(const float* __restrict__ vb,
                                          int ix, int iy, int H, int W, int dch)
{
    if (ix < 0 || ix >= W || iy < 0 || iy >= H) return 0.0f;
    long idx = (long)iy * W + ix;
    return vb[idx * 256 + dch];   // pixel stride = M*D = 256 floats
}

// One block per (b,q); wave m (of 8) handles head m; lane = D channel (D == 32 == wave32).
// Output written as bf16 so the final GEMM consumes it directly.
__global__ __launch_bounds__(256) void ms_sample(
    const float* __restrict__ value, const float* __restrict__ sl,
    const float* __restrict__ attn, __bf16* __restrict__ out,
    const int* __restrict__ Hp, const int* __restrict__ Wp, int Nq)
{
    const int H = *Hp, W = *Wp;
    const long bq = blockIdx.x;            // 0 .. b*Nq-1
    const int  m   = threadIdx.x >> 5;
    const int  dch = threadIdx.x & 31;
    const long bi = bq / Nq;
    const float* slp = sl   + bq * 64 + m * 8;   // P*2 floats per head
    const float* ap  = attn + bq * 32 + m * 4;
    const float* vb  = value + (bi * (long)(H * W)) * 256 + m * 32;

    float acc = 0.0f;
#pragma unroll
    for (int p = 0; p < 4; ++p) {
        float x = slp[2 * p]     * (float)W - 0.5f;
        float y = slp[2 * p + 1] * (float)H - 0.5f;
        float x0f = floorf(x), y0f = floorf(y);
        float lx = x - x0f, ly = y - y0f;
        int x0 = (int)x0f, y0 = (int)y0f;
        float w00 = (1.0f - lx) * (1.0f - ly);
        float w10 = lx * (1.0f - ly);
        float w01 = (1.0f - lx) * ly;
        float w11 = lx * ly;
        float s = w00 * corner_val(vb, x0,     y0,     H, W, dch)
                + w10 * corner_val(vb, x0 + 1, y0,     H, W, dch)
                + w01 * corner_val(vb, x0,     y0 + 1, H, W, dch)
                + w11 * corner_val(vb, x0 + 1, y0 + 1, H, W, dch);
        acc += ap[p] * s;
    }
    out[bq * 256 + m * 32 + dch] = f2bf(acc);
}

extern "C" void kernel_launch(void* const* d_in, const int* in_sizes, int n_in,
                              void* d_out, int out_size, void* d_ws, size_t ws_size,
                              hipStream_t stream)
{
    (void)n_in; (void)out_size; (void)ws_size;
    const float* query = (const float*)d_in[2];
    const float* ref   = (const float*)d_in[3];
    const float* Wv    = (const float*)d_in[4];
    const float* bv    = (const float*)d_in[5];
    const float* Woff  = (const float*)d_in[6];
    const float* boff  = (const float*)d_in[7];
    const float* Wa    = (const float*)d_in[8];
    const float* ba    = (const float*)d_in[9];
    const float* Wo    = (const float*)d_in[10];
    const float* bo    = (const float*)d_in[11];
    const int*   Hp    = (const int*)d_in[12];
    const int*   Wp    = (const int*)d_in[13];

    const int  b    = in_sizes[0] / (64 * 256);       // context is (b, 64, 256)
    const long Mtot = (long)in_sizes[2] / 256;        // b * Nq
    const int  Nq   = (int)(Mtot / b);

    float* out   = (float*)d_out;                 // (b, Nq, 256)
    float* sl    = out + Mtot * 256;              // (b, Nq, M, 1, P, 2) = 64/row
    float* attnp = sl + Mtot * 64;                // (b, Nq, M, P)       = 32/row

    // workspace layout (all 16B aligned)
    char* ws = (char*)d_ws;
    __bf16* qbf     = (__bf16*)ws;                 ws += Mtot * 256 * sizeof(__bf16);
    __bf16* sampled = (__bf16*)ws;                 ws += Mtot * 256 * sizeof(__bf16);
    float*  value   = (float*)ws;                  ws += Mtot * 256 * sizeof(float);
    __bf16* WvT     = (__bf16*)ws;                 ws += 256 * 256 * sizeof(__bf16);
    __bf16* WoT     = (__bf16*)ws;                 ws += 256 * 256 * sizeof(__bf16);
    __bf16* WoffT   = (__bf16*)ws;                 ws += 256 * 64 * sizeof(__bf16);
    __bf16* WaT     = (__bf16*)ws;                 ws += 256 * 32 * sizeof(__bf16);

    dim3 blk(256);
    const unsigned gx = (unsigned)(Mtot / 512);

    // one-shot conversions
    {
        long n = Mtot * 256;
        cvt_bf16<<<(unsigned)((n / 4 + 255) / 256), blk, 0, stream>>>(query, qbf, n);
    }
    cvt_transpose<<<(256 * 256 + 255) / 256, blk, 0, stream>>>(Wv,   WvT,   256);
    cvt_transpose<<<(256 * 256 + 255) / 256, blk, 0, stream>>>(Wo,   WoT,   256);
    cvt_transpose<<<(256 * 64  + 255) / 256, blk, 0, stream>>>(Woff, WoffT, 64);
    cvt_transpose<<<(256 * 32  + 255) / 256, blk, 0, stream>>>(Wa,   WaT,   32);

    // offsets -> sl region, then transform in place to sampling locations
    gemm_bf16_wmma<<<dim3(gx, 64 / 16), blk, 0, stream>>>(qbf, WoffT, boff, sl, 64);
    {
        long tot = Mtot * 64;
        loc_fix<<<(unsigned)((tot + 255) / 256), blk, 0, stream>>>(sl, ref, Hp, Wp, tot);
    }
    // attention logits -> attn region, softmax in place over P=4
    gemm_bf16_wmma<<<dim3(gx, 32 / 16), blk, 0, stream>>>(qbf, WaT, ba, attnp, 32);
    {
        long groups = Mtot * 8;   // b*Nq*M
        softmax4<<<(unsigned)((groups + 255) / 256), blk, 0, stream>>>(attnp, groups);
    }
    // value projection (f32 output for precise gathers; fits in 192MB L2)
    gemm_bf16_wmma<<<dim3(gx, 256 / 16), blk, 0, stream>>>(qbf, WvT, bv, value, 256);
    // deformable bilinear sampling -> bf16 directly
    ms_sample<<<(unsigned)Mtot, blk, 0, stream>>>(value, sl, attnp, sampled, Hp, Wp, Nq);
    // output projection
    gemm_bf16_wmma<<<dim3(gx, 256 / 16), blk, 0, stream>>>(sampled, WoT, bo, out, 256);
}
